// SNN_49443663511798
// MI455X (gfx1250) — matrix operand order repaired and checked by reading, not compiled, not run on hardware
//
#include <hip/hip_runtime.h>

// ---------------------------------------------------------------------------
// SNN: 3x (Linear -> LIF) ; big GEMM done once for all T, LIF scans are tiny.
// bf16 WMMA (v_wmma_f32_16x16x32_bf16), fp32 accumulate, deterministic K-split.
// ---------------------------------------------------------------------------

typedef __attribute__((ext_vector_type(16))) __bf16 v16bf;
typedef __attribute__((ext_vector_type(8)))  __bf16 v8bf;
typedef __attribute__((ext_vector_type(8)))  float  v8f;

#define TT      16
#define BB      64
#define MROWS   1024          // T*B
#define K1      60000
#define LD1     64            // padded N of layer1 (actual 50)
#define LD2     256           // padded N of layer2 (actual 200)
#define LD3     16            // padded N of layer3 (actual 2)
#define KC      160           // K-chunk for gemm1 (5 wmma steps)
#define NSEG    15            // K-split segments (deterministic partials)
#define KSEG    4000          // K1 / NSEG
#define NCHUNK  25            // KSEG / KC

// workspace layout (bytes, all 32B aligned)
static constexpr size_t OFF_W1P = 0;                       // 64*60000*2   = 7,680,000
static constexpr size_t OFF_W2P = 7680000;                 // 256*64*2     = 32,768
static constexpr size_t OFF_W3P = OFF_W2P + 32768;         // 16*256*2     = 8,192
static constexpr size_t OFF_H1P = OFF_W3P + 8192;          // 15*1024*64*4 = 3,932,160
static constexpr size_t OFF_S1P = OFF_H1P + 3932160;       // 1024*64*2    = 131,072
static constexpr size_t OFF_H2  = OFF_S1P + 131072;        // 1024*256*4   = 1,048,576
static constexpr size_t OFF_S2P = OFF_H2  + 1048576;       // 1024*256*2   = 524,288
static constexpr size_t OFF_H3  = OFF_S2P + 524288;        // 1024*16*4    = 65,536

static __device__ __forceinline__ v16bf make_afrag(v8bf lo, v8bf hi) {
  return __builtin_shufflevector(lo, hi, 0,1,2,3,4,5,6,7,8,9,10,11,12,13,14,15);
}

// ---------------------------------------------------------------------------
// Weight conversion fp32 -> bf16, zero-padded so padded rows/cols contribute 0.
// ---------------------------------------------------------------------------
__global__ void snn_convert_weights(const float* __restrict__ W1,
                                    const float* __restrict__ W2,
                                    const float* __restrict__ W3,
                                    __bf16* __restrict__ w1p,
                                    __bf16* __restrict__ w2p,
                                    __bf16* __restrict__ w3p) {
  const long n1 = 64L * 60000, n2 = 256L * 64, n3 = 16L * 256;
  const long total = n1 + n2 + n3;
  for (long i = blockIdx.x * (long)blockDim.x + threadIdx.x; i < total;
       i += (long)gridDim.x * blockDim.x) {
    if (i < n1) {
      long o = i / 60000, k = i % 60000;
      w1p[i] = (o < 50) ? (__bf16)W1[o * 60000 + k] : (__bf16)0.f;
    } else if (i < n1 + n2) {
      long j = i - n1, o = j / 64, k = j % 64;
      w2p[j] = (o < 200 && k < 50) ? (__bf16)W2[o * 50 + k] : (__bf16)0.f;
    } else {
      long j = i - n1 - n2, o = j / 256, k = j % 256;
      w3p[j] = (o < 2 && k < 200) ? (__bf16)W3[o * 200 + k] : (__bf16)0.f;
    }
  }
}

// ---------------------------------------------------------------------------
// GEMM1: h1part[seg] += x[16 rows] * W1^T, bf16 WMMA, x staged once in LDS
// grid = (64 M-tiles, 15 K-segments), 128 threads = 4 waves (one N-tile each)
// ---------------------------------------------------------------------------
__global__ __launch_bounds__(128)
void snn_gemm1(const float* __restrict__ x, const __bf16* __restrict__ w1p,
               float* __restrict__ h1part) {
  __shared__ __bf16 As[16 * KC];                 // 5 KB of 320 KB/WGP
  const int mtile = blockIdx.x;                  // 0..63
  const int seg   = blockIdx.y;                  // 0..14
  const int tid   = threadIdx.x;
  const int wave  = tid >> 5;                    // N-tile 0..3
  const int lane  = tid & 31;
  const int half  = lane >> 4;
  const int l15   = lane & 15;

  const float*  xblk = x   + (size_t)(mtile * 16) * K1 + (size_t)seg * KSEG;
  const __bf16* brow = w1p + (size_t)(wave * 16 + l15) * K1 + (size_t)seg * KSEG;

  v8f acc = {};
  for (int ch = 0; ch < NCHUNK; ++ch) {
    const int kbase = ch * KC;
    // cooperative load: 16 x 160 fp32 -> bf16 LDS (x read exactly once)
    #pragma unroll
    for (int it = 0; it < 5; ++it) {
      int f4 = it * 128 + tid;                   // 0..639 float4 slots
      int r  = f4 / (KC / 4);
      int c4 = (f4 % (KC / 4)) * 4;
      float4 v = *(const float4*)(xblk + (size_t)r * K1 + kbase + c4);
      __bf16* dst = &As[r * KC + c4];
      dst[0] = (__bf16)v.x; dst[1] = (__bf16)v.y;
      dst[2] = (__bf16)v.z; dst[3] = (__bf16)v.w;
    }
    if (ch + 1 < NCHUNK)                         // global_prefetch next chunk
      __builtin_prefetch(xblk + (size_t)(tid & 15) * K1 + kbase + KC, 0, 0);
    __syncthreads();
    #pragma unroll
    for (int kk = 0; kk < KC; kk += 32) {
      // A 16x32 bf16: lane holds M=l15, K = 16*half+{0..15}
      v8bf alo = *(const v8bf*)&As[l15 * KC + kk + 16 * half];
      v8bf ahi = *(const v8bf*)&As[l15 * KC + kk + 16 * half + 8];
      v16bf a  = make_afrag(alo, ahi);
      // B 32x16 bf16: lane holds N=l15, K = 16*half..+16 ; B(k,n) = W1[n][k]
      v16bf b  = *(const v16bf*)(brow + kbase + kk + 16 * half);
      acc = __builtin_amdgcn_wmma_f32_16x16x32_bf16(
          false, a, false, b, (short)0, acc, false, false);
    }
    __syncthreads();
  }
  // C layout: VGPR i, lanes0-15 -> M=i ; lanes16-31 -> M=8+i ; N = l15
  float* out = h1part + (size_t)seg * MROWS * LD1;
  const int n = wave * 16 + l15;
  const int mbase = mtile * 16 + (half ? 8 : 0);
  #pragma unroll
  for (int i = 0; i < 8; ++i)
    out[(size_t)(mbase + i) * LD1 + n] = acc[i];
}

// ---------------------------------------------------------------------------
// Generic small bf16 GEMM: one wave per 16x16 C tile, operands from L2.
// ---------------------------------------------------------------------------
__global__ __launch_bounds__(32)
void snn_gemm_bf16(const __bf16* __restrict__ A, int lda,
                   const __bf16* __restrict__ W, int ldb,
                   float* __restrict__ C, int ldc, int ksteps) {
  const int mtile = blockIdx.x, ntile = blockIdx.y;
  const int lane = threadIdx.x & 31;
  const int half = lane >> 4, l15 = lane & 15;
  const __bf16* arow = A + (size_t)(mtile * 16 + l15) * lda;
  const __bf16* brow = W + (size_t)(ntile * 16 + l15) * ldb;
  v8f acc = {};
  for (int ks = 0; ks < ksteps; ++ks) {
    const int k0 = ks * 32;
    v8bf alo = *(const v8bf*)(arow + k0 + 16 * half);
    v8bf ahi = *(const v8bf*)(arow + k0 + 16 * half + 8);
    v16bf a  = make_afrag(alo, ahi);
    v16bf b  = *(const v16bf*)(brow + k0 + 16 * half);
    acc = __builtin_amdgcn_wmma_f32_16x16x32_bf16(
        false, a, false, b, (short)0, acc, false, false);
  }
  const int n = ntile * 16 + l15;
  const int mbase = mtile * 16 + (half ? 8 : 0);
  #pragma unroll
  for (int i = 0; i < 8; ++i)
    C[(size_t)(mbase + i) * ldc + n] = acc[i];
}

// ---------------------------------------------------------------------------
// LIF scan over T (v += (x-v)/2 ; spike = v>=1 ; hard reset), writes padded
// bf16 spikes. nseg>1 deterministically reduces K-split partials of gemm1.
// ---------------------------------------------------------------------------
__global__ void snn_lif_pad(const float* __restrict__ hin, int ld, int nseg,
                            long segstride, __bf16* __restrict__ sout, int D) {
  const int tid = blockIdx.x * blockDim.x + threadIdx.x;
  if (tid >= BB * ld) return;
  const int b = tid / ld, o = tid % ld;
  if (o >= D) {                                   // zero the pad columns
    for (int t = 0; t < TT; ++t)
      sout[(size_t)(t * BB + b) * ld + o] = (__bf16)0.f;
    return;
  }
  float v = 0.f;
  for (int t = 0; t < TT; ++t) {
    const size_t idx = (size_t)(t * BB + b) * ld + o;
    float xin = 0.f;
    for (int s = 0; s < nseg; ++s) xin += hin[idx + (size_t)s * segstride];
    v += (xin - v) * 0.5f;
    const float spk = (v >= 1.0f) ? 1.f : 0.f;
    v *= (1.f - spk);
    sout[idx] = (__bf16)spk;
  }
}

__global__ void snn_lif_out(const float* __restrict__ hin,
                            float* __restrict__ out) {
  const int tid = blockIdx.x * blockDim.x + threadIdx.x;
  if (tid >= BB * 2) return;
  const int b = tid / 2, o = tid % 2;
  float v = 0.f;
  for (int t = 0; t < TT; ++t) {
    const float xin = hin[(size_t)(t * BB + b) * LD3 + o];
    v += (xin - v) * 0.5f;
    const float spk = (v >= 1.0f) ? 1.f : 0.f;
    v *= (1.f - spk);
    out[(t * BB + b) * 2 + o] = spk;              // [T, B, 2] fp32
  }
}

// ---------------------------------------------------------------------------
extern "C" void kernel_launch(void* const* d_in, const int* in_sizes, int n_in,
                              void* d_out, int out_size, void* d_ws, size_t ws_size,
                              hipStream_t stream) {
  const float* x  = (const float*)d_in[0];   // [16,64,150,400] = [1024][60000]
  const float* W1 = (const float*)d_in[1];   // [50][60000]
  const float* W2 = (const float*)d_in[2];   // [200][50]
  const float* W3 = (const float*)d_in[3];   // [2][200]

  char* ws = (char*)d_ws;
  __bf16* w1p   = (__bf16*)(ws + OFF_W1P);
  __bf16* w2p   = (__bf16*)(ws + OFF_W2P);
  __bf16* w3p   = (__bf16*)(ws + OFF_W3P);
  float*  h1p   = (float*) (ws + OFF_H1P);   // [15][1024][64]
  __bf16* s1p   = (__bf16*)(ws + OFF_S1P);   // [1024][64]
  float*  h2    = (float*) (ws + OFF_H2);    // [1024][256]
  __bf16* s2p   = (__bf16*)(ws + OFF_S2P);   // [1024][256]
  float*  h3    = (float*) (ws + OFF_H3);    // [1024][16]

  snn_convert_weights<<<1024, 256, 0, stream>>>(W1, W2, W3, w1p, w2p, w3p);

  // Layer 1: GEMM (M=1024, K=60000, N=64pad), 64 M-tiles x 15 K-segments
  snn_gemm1<<<dim3(64, NSEG), 128, 0, stream>>>(x, w1p, h1p);
  snn_lif_pad<<<(BB * LD1 + 255) / 256, 256, 0, stream>>>(
      h1p, LD1, NSEG, (long)MROWS * LD1, s1p, 50);

  // Layer 2: M=1024, K=64pad (2 wmma steps), N tiles 0..12 cover 208 >= 200
  snn_gemm_bf16<<<dim3(64, 13), 32, 0, stream>>>(s1p, LD1, w2p, LD1, h2, LD2, 2);
  snn_lif_pad<<<(BB * LD2 + 255) / 256, 256, 0, stream>>>(
      h2, LD2, 1, 0, s2p, 200);

  // Layer 3: M=1024, K=256pad (8 wmma steps), single N tile
  snn_gemm_bf16<<<dim3(64, 1), 32, 0, stream>>>(s2p, LD2, w3p, LD2, h3, LD3, 8);
  snn_lif_out<<<1, 128, 0, stream>>>(h3, (float*)d_out);
}
